// TemporalMambaBlock_76338748719295
// MI455X (gfx1250) — matrix operand order repaired and checked by reading, not compiled, not run on hardware
//
#include <hip/hip_runtime.h>
#include <hip/hip_bf16.h>

// ---------------- problem constants ----------------
constexpr int Bb   = 2;
constexpr int Ls   = 1024;
constexpr int Dd   = 1024;
constexpr int DIc  = 2048;
constexpr int DSc  = 16;
constexpr int DTRc = 64;
constexpr int Mrows = Bb * Ls;          // 2048
constexpr int XDBL  = DTRc + 2 * DSc;   // 96

typedef __bf16 v16bf __attribute__((ext_vector_type(16)));
typedef float  v8f   __attribute__((ext_vector_type(8)));

union Frag { v16bf v; uint4 q[2]; };

static __device__ __forceinline__ unsigned short f2bf(float f) {
  unsigned int u = __float_as_uint(f);
  return (unsigned short)((u + 0x7FFFu + ((u >> 16) & 1u)) >> 16);
}

// ---------------- elementwise kernels ----------------
__global__ void cast_f32_bf16(const float* __restrict__ s, unsigned short* __restrict__ d, int n) {
  int i = blockIdx.x * blockDim.x + threadIdx.x;
  if (i < n) d[i] = f2bf(s[i]);
}

__global__ void rmsnorm_bf16(const float* __restrict__ x, const float* __restrict__ w,
                             unsigned short* __restrict__ o, int D) {
  int row = blockIdx.x;
  const float* xr = x + (size_t)row * D;
  float ss = 0.f;
  for (int i = threadIdx.x; i < D; i += blockDim.x) { float v = xr[i]; ss += v * v; }
  __shared__ float red[256];
  red[threadIdx.x] = ss; __syncthreads();
  for (int s = 128; s > 0; s >>= 1) {
    if ((int)threadIdx.x < s) red[threadIdx.x] += red[threadIdx.x + s];
    __syncthreads();
  }
  float inv = rsqrtf(red[0] / (float)D + 1e-6f);
  unsigned short* orow = o + (size_t)row * D;
  for (int i = threadIdx.x; i < D; i += blockDim.x) orow[i] = f2bf(w[i] * xr[i] * inv);
}

__global__ void conv_silu(const float* __restrict__ xz, const float* __restrict__ cw,
                          const float* __restrict__ cb, float* __restrict__ xo,
                          unsigned short* __restrict__ xob) {
  int l = blockIdx.x, b = blockIdx.y;
  for (int d = threadIdx.x; d < DIc; d += blockDim.x) {
    float acc = cb[d];
#pragma unroll
    for (int k = 0; k < 4; ++k) {
      int ll = l - 3 + k;
      if (ll >= 0) acc += cw[d * 4 + k] * xz[((size_t)b * Ls + ll) * (2 * DIc) + d];
    }
    float s = acc / (1.f + __expf(-acc));
    size_t oi = ((size_t)b * Ls + l) * DIc + d;
    xo[oi] = s; xob[oi] = f2bf(s);
  }
}

__global__ void cast_dtin(const float* __restrict__ xdbl, unsigned short* __restrict__ o) {
  int i = blockIdx.x * blockDim.x + threadIdx.x;
  if (i < Mrows * DTRc) {
    int r = i >> 6, c = i & 63;
    o[i] = f2bf(xdbl[r * XDBL + c]);
  }
}

__global__ void softplus_bias(float* __restrict__ dt, const float* __restrict__ bias, int n) {
  int i = blockIdx.x * blockDim.x + threadIdx.x;
  if (i < n) {
    float v = dt[i] + bias[i % DIc];
    dt[i] = (v > 20.f) ? v : log1pf(__expf(v));
  }
}

__global__ void add2(const float* __restrict__ a, const float* __restrict__ b,
                     float* __restrict__ o, int n) {
  int i = blockIdx.x * blockDim.x + threadIdx.x;
  if (i < n) o[i] = a[i] + b[i];
}

__global__ void mlp_act(const float* __restrict__ g, const float* __restrict__ bias,
                        unsigned short* __restrict__ o, int n, int nb) {
  int i = blockIdx.x * blockDim.x + threadIdx.x;
  if (i < n) {
    float v = g[i] + bias[i % nb];
    o[i] = f2bf(v / (1.f + __expf(-v)));
  }
}

__global__ void final_add(const float* __restrict__ h2, const float* __restrict__ raw,
                          const float* __restrict__ bias, float* __restrict__ o, int n, int nb) {
  int i = blockIdx.x * blockDim.x + threadIdx.x;
  if (i < n) o[i] = h2[i] + raw[i] + bias[i % nb];
}

// ---------------- selective-scan kernel (serial over L, parallel over B*DI) ----------------
__global__ __launch_bounds__(256) void ssm_scan(
    const float* __restrict__ dt, const float* __restrict__ xc,
    const float* __restrict__ xdbl, const float* __restrict__ xz,
    const float* __restrict__ Alog, const float* __restrict__ Dp,
    unsigned short* __restrict__ ybf) {
  int d = blockIdx.x * blockDim.x + threadIdx.x;
  int b = blockIdx.y;
  float As[DSc], h[DSc];
#pragma unroll
  for (int s = 0; s < DSc; ++s) { As[s] = -__expf(Alog[d * DSc + s]); h[s] = 0.f; }
  float Dpar = Dp[d];
  for (int l = 0; l < Ls; ++l) {
    size_t row = (size_t)b * Ls + l;
    float dtv = dt[row * DIc + d];
    float xv  = xc[row * DIc + d];
    const float* bc = xdbl + row * XDBL;
    float dtx = dtv * xv;
    float y = 0.f;
#pragma unroll
    for (int s = 0; s < DSc; ++s) {
      float dA = __expf(dtv * As[s]);
      h[s] = dA * h[s] + dtx * bc[DTRc + s];
      y += h[s] * bc[DTRc + DSc + s];
    }
    y += Dpar * xv;
    float zv = xz[row * (2 * DIc) + DIc + d];
    y *= zv / (1.f + __expf(-zv));
    ybf[row * DIc + d] = f2bf(y);
  }
}

// ---------------- WMMA bf16 GEMM: O[M,N] = A[M,K] * W[N,K]^T ----------------
// Block tile 128x128, 8 waves of 64x32, BK=32 (= one v_wmma_f32_16x16x32_bf16 K-step),
// double-buffered LDS, global_prefetch for the tile after next.
constexpr int BM = 128, BN = 128, BK = 32, LDA = 48;  // LDA pad -> 96B rows, 16B-aligned frags

__global__ __launch_bounds__(256) void gemm_bf16_wmma(
    const unsigned short* __restrict__ A, const unsigned short* __restrict__ W,
    float* __restrict__ O, int M, int N, int K) {
  __shared__ __align__(16) unsigned short As[2][BM * LDA];
  __shared__ __align__(16) unsigned short Bs[2][BN * LDA];

  const int tid  = threadIdx.x;
  const int lane = tid & 31;
  const int wid  = tid >> 5;
  const int wm   = (wid >> 2) * 64;   // 2 waves along M
  const int wn   = (wid & 3) * 32;    // 4 waves along N
  const int lm   = lane & 15;
  const int lh   = lane >> 4;

  const int m0 = blockIdx.y * BM;
  const int n0 = blockIdx.x * BN;

  // each thread stages 16 bf16 (2x b128) of A-tile and of B-tile per buffer
  const int sr = tid >> 1;            // staging row 0..127
  const int sc = (tid & 1) * 16;      // staging k-offset (elements)

  uint4 arg[2], brg[2];
  const int ktiles = K / BK;

  auto ldglobal = [&](int k0) {
    if (m0 + sr < M) {
      const uint4* p = (const uint4*)(A + (size_t)(m0 + sr) * K + k0 + sc);
      arg[0] = p[0]; arg[1] = p[1];
    } else {
      arg[0] = make_uint4(0, 0, 0, 0); arg[1] = make_uint4(0, 0, 0, 0);
    }
    if (n0 + sr < N) {
      const uint4* p = (const uint4*)(W + (size_t)(n0 + sr) * K + k0 + sc);
      brg[0] = p[0]; brg[1] = p[1];
    } else {
      brg[0] = make_uint4(0, 0, 0, 0); brg[1] = make_uint4(0, 0, 0, 0);
    }
  };
  auto stlds = [&](int buf) {
    *(uint4*)&As[buf][sr * LDA + sc]     = arg[0];
    *(uint4*)&As[buf][sr * LDA + sc + 8] = arg[1];
    *(uint4*)&Bs[buf][sr * LDA + sc]     = brg[0];
    *(uint4*)&Bs[buf][sr * LDA + sc + 8] = brg[1];
  };

  v8f acc[4][2];
#pragma unroll
  for (int i = 0; i < 4; ++i)
#pragma unroll
    for (int j = 0; j < 2; ++j)
#pragma unroll
      for (int r = 0; r < 8; ++r) acc[i][j][r] = 0.f;

  auto compute = [&](int buf) {
    Frag fa[4], fb[2];
#pragma unroll
    for (int i = 0; i < 4; ++i) {
      // A-frag: lane m = lm, K pattern {0..7,16..23} (lanes 0-15) / {8..15,24..31}
      const unsigned short* p = &As[buf][(wm + i * 16 + lm) * LDA + lh * 8];
      fa[i].q[0] = *(const uint4*)p;
      fa[i].q[1] = *(const uint4*)(p + 16);
    }
#pragma unroll
    for (int j = 0; j < 2; ++j) {
      // B-frag: lane n = lm, K pattern {0..7,16..23} (lanes 0-15) / {8..15,24..31}
      const unsigned short* p = &Bs[buf][(wn + j * 16 + lm) * LDA + lh * 8];
      fb[j].q[0] = *(const uint4*)p;
      fb[j].q[1] = *(const uint4*)(p + 16);
    }
#pragma unroll
    for (int i = 0; i < 4; ++i)
#pragma unroll
      for (int j = 0; j < 2; ++j)
        acc[i][j] = __builtin_amdgcn_wmma_f32_16x16x32_bf16(
            false, fa[i].v, false, fb[j].v, (short)0, acc[i][j], false, false);
  };

  ldglobal(0);
  stlds(0);
  __syncthreads();

  int cur = 0;
  for (int kt = 0; kt < ktiles; ++kt) {
    if (kt + 1 < ktiles) {
      ldglobal((kt + 1) * BK);
      if (kt + 2 < ktiles) {  // global_prefetch_b8 for the tile after next
        int k2 = (kt + 2) * BK;
        if (m0 + sr < M) __builtin_prefetch(A + (size_t)(m0 + sr) * K + k2 + sc, 0, 1);
        if (n0 + sr < N) __builtin_prefetch(W + (size_t)(n0 + sr) * K + k2 + sc, 0, 1);
      }
    }
    compute(cur);
    if (kt + 1 < ktiles) {
      stlds(cur ^ 1);
      __syncthreads();
    }
    cur ^= 1;
  }

  // D store: VGPR r holds M = r + 8*lh (16x16 f32 C/D layout)
#pragma unroll
  for (int i = 0; i < 4; ++i) {
#pragma unroll
    for (int j = 0; j < 2; ++j) {
      int n = n0 + wn + j * 16 + lm;
      if (n < N) {
        size_t mb = (size_t)(m0 + wm + i * 16 + lh * 8);
        float* po = O + mb * (size_t)N + n;
#pragma unroll
        for (int r = 0; r < 8; ++r) po[(size_t)r * N] = acc[i][j][r];
      }
    }
  }
}

// ---------------- host-side orchestration ----------------
static inline int ceildiv(int a, int b) { return (a + b - 1) / b; }

extern "C" void kernel_launch(void* const* d_in, const int* in_sizes, int n_in,
                              void* d_out, int out_size, void* d_ws, size_t ws_size,
                              hipStream_t stream) {
  (void)in_sizes; (void)n_in; (void)out_size; (void)ws_size;
  const float* hidden    = (const float*)d_in[0];
  const float* norm1_w   = (const float*)d_in[1];
  const float* in_proj_w = (const float*)d_in[2];   // (4096,1024)
  const float* conv_w    = (const float*)d_in[3];   // (2048,4)
  const float* conv_b    = (const float*)d_in[4];
  const float* x_proj_w  = (const float*)d_in[5];   // (96,2048)
  const float* dt_proj_w = (const float*)d_in[6];   // (2048,64)
  const float* dt_proj_b = (const float*)d_in[7];
  const float* A_log     = (const float*)d_in[8];   // (2048,16)
  const float* D_param   = (const float*)d_in[9];
  const float* out_proj_w= (const float*)d_in[10];  // (1024,2048)
  const float* norm2_w   = (const float*)d_in[11];
  const float* fc1_w     = (const float*)d_in[12];  // (4096,1024)
  const float* fc1_b     = (const float*)d_in[13];
  const float* fc2_w     = (const float*)d_in[14];  // (1024,4096)
  const float* fc2_b     = (const float*)d_in[15];
  float* out = (float*)d_out;

  // workspace carve-up (256B aligned)
  char* ws = (char*)d_ws;
  size_t off = 0;
  auto alloc = [&](size_t bytes) { char* p = ws + off; off = (off + bytes + 255) & ~(size_t)255; return p; };

  unsigned short* wb_in  = (unsigned short*)alloc((size_t)2 * DIc * Dd * 2);       // 4096x1024
  unsigned short* wb_xp  = (unsigned short*)alloc((size_t)XDBL * DIc * 2);         // 96x2048
  unsigned short* wb_dt  = (unsigned short*)alloc((size_t)DIc * DTRc * 2);         // 2048x64
  unsigned short* wb_out = (unsigned short*)alloc((size_t)Dd * DIc * 2);           // 1024x2048
  unsigned short* wb_fc1 = (unsigned short*)alloc((size_t)4 * Dd * Dd * 2);        // 4096x1024
  unsigned short* wb_fc2 = (unsigned short*)alloc((size_t)Dd * 4 * Dd * 2);        // 1024x4096
  unsigned short* hn     = (unsigned short*)alloc((size_t)Mrows * Dd * 2);         // norm out (reused)
  float*          G      = (float*)alloc((size_t)Mrows * 4 * Dd * 4);              // xz, later fc1 raw
  float*          xconv  = (float*)alloc((size_t)Mrows * DIc * 4);
  unsigned short* xconvb = (unsigned short*)alloc((size_t)Mrows * DIc * 2);
  float*          xdbl   = (float*)alloc((size_t)Mrows * XDBL * 4);
  unsigned short* dtin   = (unsigned short*)alloc((size_t)Mrows * DTRc * 2);
  float*          dtf    = (float*)alloc((size_t)Mrows * DIc * 4);
  unsigned short* ybf    = (unsigned short*)alloc((size_t)Mrows * DIc * 2);
  unsigned short* mbf    = (unsigned short*)alloc((size_t)Mrows * 4 * Dd * 2);
  float*          smallo = (float*)alloc((size_t)Mrows * Dd * 4);                  // out_proj / fc2 raw
  float*          h2     = (float*)alloc((size_t)Mrows * Dd * 4);

  const int T = 256;
  // 1) weights -> bf16
  { int n = 2*DIc*Dd;  cast_f32_bf16<<<ceildiv(n,T), T, 0, stream>>>(in_proj_w,  wb_in,  n); }
  { int n = XDBL*DIc;  cast_f32_bf16<<<ceildiv(n,T), T, 0, stream>>>(x_proj_w,   wb_xp,  n); }
  { int n = DIc*DTRc;  cast_f32_bf16<<<ceildiv(n,T), T, 0, stream>>>(dt_proj_w,  wb_dt,  n); }
  { int n = Dd*DIc;    cast_f32_bf16<<<ceildiv(n,T), T, 0, stream>>>(out_proj_w, wb_out, n); }
  { int n = 4*Dd*Dd;   cast_f32_bf16<<<ceildiv(n,T), T, 0, stream>>>(fc1_w,      wb_fc1, n); }
  { int n = Dd*4*Dd;   cast_f32_bf16<<<ceildiv(n,T), T, 0, stream>>>(fc2_w,      wb_fc2, n); }

  // 2) RMSNorm1
  rmsnorm_bf16<<<Mrows, T, 0, stream>>>(hidden, norm1_w, hn, Dd);
  // 3) in_proj: (2048 x 4096) = hn (2048x1024) * in_proj_w^T
  gemm_bf16_wmma<<<dim3(ceildiv(2*DIc,BN), Mrows/BM), T, 0, stream>>>(hn, wb_in, G, Mrows, 2*DIc, Dd);
  // 4) causal conv + silu
  conv_silu<<<dim3(Ls, Bb), T, 0, stream>>>(G, conv_w, conv_b, xconv, xconvb);
  // 5) x_proj: (2048 x 96)
  gemm_bf16_wmma<<<dim3(ceildiv(XDBL,BN), Mrows/BM), T, 0, stream>>>(xconvb, wb_xp, xdbl, Mrows, XDBL, DIc);
  // 6) dt_proj: (2048 x 2048), K=64
  { int n = Mrows*DTRc; cast_dtin<<<ceildiv(n,T), T, 0, stream>>>(xdbl, dtin); }
  gemm_bf16_wmma<<<dim3(ceildiv(DIc,BN), Mrows/BM), T, 0, stream>>>(dtin, wb_dt, dtf, Mrows, DIc, DTRc);
  { int n = Mrows*DIc; softplus_bias<<<ceildiv(n,T), T, 0, stream>>>(dtf, dt_proj_b, n); }
  // 7) selective scan + gate
  ssm_scan<<<dim3(DIc/T, Bb), T, 0, stream>>>(dtf, xconv, xdbl, G, A_log, D_param, ybf);
  // 8) out_proj + residual
  gemm_bf16_wmma<<<dim3(ceildiv(Dd,BN), Mrows/BM), T, 0, stream>>>(ybf, wb_out, smallo, Mrows, Dd, DIc);
  { int n = Mrows*Dd; add2<<<ceildiv(n,T), T, 0, stream>>>(hidden, smallo, h2, n); }
  // 9) RMSNorm2
  rmsnorm_bf16<<<Mrows, T, 0, stream>>>(h2, norm2_w, hn, Dd);
  // 10) fc1 + silu
  gemm_bf16_wmma<<<dim3(ceildiv(4*Dd,BN), Mrows/BM), T, 0, stream>>>(hn, wb_fc1, G, Mrows, 4*Dd, Dd);
  { int n = Mrows*4*Dd; mlp_act<<<ceildiv(n,T), T, 0, stream>>>(G, fc1_b, mbf, n, 4*Dd); }
  // 11) fc2 + bias + residual -> out
  gemm_bf16_wmma<<<dim3(ceildiv(Dd,BN), Mrows/BM), T, 0, stream>>>(mbf, wb_fc2, smallo, Mrows, Dd, 4*Dd);
  { int n = Mrows*Dd; final_add<<<ceildiv(n,T), T, 0, stream>>>(h2, smallo, fc2_b, out, n, Dd); }
}
